// KCN_38955353375381
// MI455X (gfx1250) — matrix operand order repaired and checked by reading, not compiled, not run on hardware
//
#include <hip/hip_runtime.h>
#include <hip/hip_bf16.h>
#include <math.h>

#define N_TOK 4096
#define DIN   512
#define DOUT  512
#define KSH   8
#define DK    (DIN * KSH)   /* 4096 */
#define NP    8

typedef __attribute__((ext_vector_type(16))) __bf16 v16bf;
typedef __attribute__((ext_vector_type(8)))  __bf16 v8bf;
typedef __attribute__((ext_vector_type(8)))  float  v8f;

static __device__ __forceinline__ float sp_f(float x) {
  return (x > 20.0f) ? x : log1pf(__expf(x));
}
static __device__ __forceinline__ float sigmoid_f(float x) {
  return 1.0f / (1.0f + __expf(-x));
}
static __device__ __forceinline__ __bf16 f2bf(float f) {
  unsigned u = __builtin_bit_cast(unsigned, f);
  unsigned r = u + 0x7FFFu + ((u >> 16) & 1u);   // round-to-nearest-even
  unsigned short h = (unsigned short)(r >> 16);
  return __builtin_bit_cast(__bf16, h);
}

// A fragment (16x32 bf16, MxK): lane holds row (lane&15); its two 8-element
// K chunks sit at +0 and +8 relative to `base` (base already includes hl*16).
static __device__ __forceinline__ v16bf load_frag_a(const __bf16* __restrict__ base) {
  v8bf lo = *(const v8bf*)(base);
  v8bf hi = *(const v8bf*)(base + 8);
  v16bf r;
#pragma unroll
  for (int i = 0; i < 8; ++i) { r[i] = lo[i]; r[i + 8] = hi[i]; }
  return r;
}

static __device__ __forceinline__ v8f wmma_bf16(v16bf a, v16bf b, v8f c) {
  return __builtin_amdgcn_wmma_f32_16x16x32_bf16(false, a, false, b,
                                                 (short)0, c, false, false);
}

// ---------------- prep kernels ----------------

__global__ void k_sp2_bf16(const float* __restrict__ src, __bf16* __restrict__ dst) {
  int i = blockIdx.x * blockDim.x + threadIdx.x;
  float s = sp_f(src[i]);
  dst[i] = f2bf(s * s);
}

// fold concat([x,x]) weight halves and transpose to [p][e][c] (c contiguous)
__global__ void k_prep_zw(const float* __restrict__ zw, __bf16* __restrict__ zwT) {
  int i = blockIdx.x * blockDim.x + threadIdx.x;   // [p][e][c]
  int c = i & (DIN - 1);
  int e = (i >> 9) & (DOUT - 1);
  int p = i >> 18;
  const float* base = zw + (size_t)p * (2 * DIN) * DOUT;
  float v = base[(size_t)c * DOUT + e] + base[(size_t)(c + DIN) * DOUT + e];
  zwT[i] = f2bf(v);
}

// x -> bf16 copy, and phi[n][c*8+k] = bf16(softplus(x + shift_k))
__global__ void k_prep_xphi(const float* __restrict__ x,
                            __bf16* __restrict__ xb,
                            __bf16* __restrict__ phi) {
  int i = blockIdx.x * blockDim.x + threadIdx.x;   // N_TOK*DIN
  float xv = x[i];
  xb[i] = f2bf(xv);
  v8bf ph;
#pragma unroll
  for (int k = 0; k < KSH; ++k) {
    float sh = -1.0f + (2.0f / 7.0f) * (float)k;
    ph[k] = f2bf(sp_f(xv + sh));
  }
  *(v8bf*)(phi + (size_t)i * KSH) = ph;
}

// ---------------- GEMM1: x_proj = phi @ w_phi^T + phi_bias ----------------
// block = 256 thr (8 waves); block owns 32 rows; wave owns 64 of 512 cols.
// Wave tile: 32(M) x 64(N), 8 accumulators, fully double-buffered fragments.
__global__ void __launch_bounds__(256)
k_gemm1(const __bf16* __restrict__ phi, const __bf16* __restrict__ wphi,
        const float* __restrict__ phi_bias, float* __restrict__ xproj) {
  const int lane  = threadIdx.x & 31;
  const int wave  = threadIdx.x >> 5;
  const int mbase = blockIdx.x * 32;
  const int nbase = wave * 64;
  const int hl  = lane >> 4;
  const int l15 = lane & 15;

  v8f acc[2][4] = {};
  const __bf16* aRow0 = phi + (size_t)(mbase + l15) * DK + hl * 16;
  const __bf16* aRow1 = aRow0 + (size_t)16 * DK;
  const __bf16* bCol  = wphi + (size_t)(nbase + l15) * DK + hl * 16;

  // prologue: fragment set for kk = 0
  v16bf aC0 = load_frag_a(aRow0);
  v16bf aC1 = load_frag_a(aRow1);
  v16bf bC[4];
#pragma unroll
  for (int j = 0; j < 4; ++j) bC[j] = *(const v16bf*)(bCol + (size_t)j * 16 * DK);

  // unroll 2 so the double-buffer rotation becomes register-role alternation
  // instead of v_mov copies
#pragma unroll 2
  for (int kk = 32; kk < DK; kk += 32) {
    __builtin_prefetch(aRow0 + kk + 256, 0, 1);
    v16bf aN0 = load_frag_a(aRow0 + kk);
    v16bf aN1 = load_frag_a(aRow1 + kk);
    v16bf bN[4];
#pragma unroll
    for (int j = 0; j < 4; ++j)
      bN[j] = *(const v16bf*)(bCol + (size_t)j * 16 * DK + kk);

#pragma unroll
    for (int j = 0; j < 4; ++j) {
      acc[0][j] = wmma_bf16(aC0, bC[j], acc[0][j]);
      acc[1][j] = wmma_bf16(aC1, bC[j], acc[1][j]);
    }
    aC0 = aN0; aC1 = aN1;
#pragma unroll
    for (int j = 0; j < 4; ++j) bC[j] = bN[j];
  }
  // epilogue compute on last fragment set
#pragma unroll
  for (int j = 0; j < 4; ++j) {
    acc[0][j] = wmma_bf16(aC0, bC[j], acc[0][j]);
    acc[1][j] = wmma_bf16(aC1, bC[j], acc[1][j]);
  }

#pragma unroll
  for (int m = 0; m < 2; ++m) {
#pragma unroll
    for (int j = 0; j < 4; ++j) {
      int e = nbase + j * 16 + l15;
      float bias = phi_bias[e];
#pragma unroll
      for (int r = 0; r < 8; ++r) {
        int n = mbase + m * 16 + hl * 8 + r;
        xproj[(size_t)n * DOUT + e] = acc[m][j][r] + bias;
      }
    }
  }
}

// ------- fused GEMM2+GEMM3 + full epilogue -------
// out[n,p,e] = sp( sp((z0@w2^T + bias2)*g2) + x@zwT ) + output_bias
// z0 = sp(x_proj * g1) built on the fly (VALU co-executes with WMMA).
// A fragments double-buffered; B loads grouped ahead of the WMMA block.
// K-loop fully unrolled (16 compile-time iterations) so the pipeline tail
// guard folds away statically.
__global__ void __launch_bounds__(256)
k_gemm23(const float* __restrict__ xproj, const __bf16* __restrict__ xb,
         const __bf16* __restrict__ w2, const __bf16* __restrict__ zwT,
         const float* __restrict__ bias2, const float* __restrict__ gate_raw2,
         const float* __restrict__ gate_raw, const float* __restrict__ obias,
         float* __restrict__ out) {
  const int p     = blockIdx.y;
  const int lane  = threadIdx.x & 31;
  const int wave  = threadIdx.x >> 5;
  const int mbase = blockIdx.x * 16;
  const int nbase = wave * 64;
  const int hl  = lane >> 4;
  const int l15 = lane & 15;

  const float g1 = sigmoid_f(gate_raw[p]);
  const float g2 = sigmoid_f(gate_raw2[p]);

  v8f acc1[4] = {};   // z0 @ w2^T
  v8f acc2[4] = {};   // x  @ zwT

  const float*  a1Row = xproj + (size_t)(mbase + l15) * DOUT + hl * 16;
  const __bf16* a2Row = xb    + (size_t)(mbase + l15) * DIN  + hl * 16;
  const __bf16* b1Col = w2  + (size_t)p * DOUT * DOUT + (size_t)(nbase + l15) * DOUT + hl * 16;
  const __bf16* b2Col = zwT + (size_t)p * DIN  * DOUT + (size_t)(nbase + l15) * DIN  + hl * 16;

  // A1 builder: load f32 x_proj chunk, z0 = sp(v*g1), convert bf16
  auto make_a1 = [&](int kk) -> v16bf {
    v8f lo = *(const v8f*)(a1Row + kk);
    v8f hi = *(const v8f*)(a1Row + kk + 8);
    v16bf a;
#pragma unroll
    for (int i = 0; i < 8; ++i) {
      a[i]     = f2bf(sp_f(lo[i] * g1));
      a[i + 8] = f2bf(sp_f(hi[i] * g1));
    }
    return a;
  };

  v16bf a1C = make_a1(0);
  v16bf a2C = load_frag_a(a2Row);

  const int NIT = DOUT / 32;   // 16
#pragma unroll
  for (int it = 0; it < NIT; ++it) {
    const int kk = it * 32;
    // group all 8 B loads so waits stagger across the WMMA block
    v16bf b1[4], b2[4];
#pragma unroll
    for (int j = 0; j < 4; ++j) {
      b1[j] = *(const v16bf*)(b1Col + (size_t)j * 16 * DOUT + kk);
      b2[j] = *(const v16bf*)(b2Col + (size_t)j * 16 * DIN  + kk);
    }
    // prefetch next A fragments (double buffer); guard is compile-time
    v16bf a1N = a1C, a2N = a2C;
    if (it + 1 < NIT) {
      a1N = make_a1(kk + 32);
      a2N = load_frag_a(a2Row + kk + 32);
    }
#pragma unroll
    for (int j = 0; j < 4; ++j) {
      acc1[j] = wmma_bf16(a1C, b1[j], acc1[j]);
      acc2[j] = wmma_bf16(a2C, b2[j], acc2[j]);
    }
    a1C = a1N; a2C = a2N;
  }

#pragma unroll
  for (int j = 0; j < 4; ++j) {
    int e = nbase + j * 16 + l15;
    float b2v = bias2[p * DOUT + e];
    float obv = obias[p * DOUT + e];
#pragma unroll
    for (int r = 0; r < 8; ++r) {
      int n = mbase + hl * 8 + r;
      float t = sp_f((acc1[j][r] + b2v) * g2);
      float f = sp_f(t + acc2[j][r]) + obv;
      out[(size_t)n * (NP * DOUT) + (size_t)p * DOUT + e] = f;
    }
  }
}

extern "C" void kernel_launch(void* const* d_in, const int* in_sizes, int n_in,
                              void* d_out, int out_size, void* d_ws, size_t ws_size,
                              hipStream_t stream) {
  const float* x         = (const float*)d_in[0];
  const float* phi_raw   = (const float*)d_in[1];
  const float* phi_bias  = (const float*)d_in[2];
  const float* raw_w2    = (const float*)d_in[3];
  const float* bias2     = (const float*)d_in[4];
  const float* gate_raw2 = (const float*)d_in[5];
  const float* z_weight  = (const float*)d_in[6];
  const float* gate_raw  = (const float*)d_in[7];
  const float* obias     = (const float*)d_in[8];
  float* out = (float*)d_out;

  char* ws = (char*)d_ws;
  __bf16* wphi = (__bf16*)ws;  ws += (size_t)DOUT * DK * 2;          // 4 MB
  __bf16* w2   = (__bf16*)ws;  ws += (size_t)NP * DOUT * DOUT * 2;   // 4 MB
  __bf16* zwT  = (__bf16*)ws;  ws += (size_t)NP * DIN * DOUT * 2;    // 4 MB
  __bf16* xb   = (__bf16*)ws;  ws += (size_t)N_TOK * DIN * 2;        // 4 MB
  __bf16* phi  = (__bf16*)ws;  ws += (size_t)N_TOK * DK * 2;         // 32 MB
  float*  xprj = (float*)ws;                                         // 8 MB

  k_sp2_bf16<<<(DOUT * DK) / 256, 256, 0, stream>>>(phi_raw, wphi);
  k_sp2_bf16<<<(NP * DOUT * DOUT) / 256, 256, 0, stream>>>(raw_w2, w2);
  k_prep_zw<<<(NP * DOUT * DIN) / 256, 256, 0, stream>>>(z_weight, zwT);
  k_prep_xphi<<<(N_TOK * DIN) / 256, 256, 0, stream>>>(x, xb, phi);

  k_gemm1<<<N_TOK / 32, 256, 0, stream>>>(phi, wphi, phi_bias, xprj);
  k_gemm23<<<dim3(N_TOK / 16, NP), 256, 0, stream>>>(
      xprj, xb, w2, zwT, bias2, gate_raw2, gate_raw, obias, out);

  (void)in_sizes; (void)n_in; (void)out_size; (void)ws_size;
}
